// GCNBertClassifier_27779848471014
// MI455X (gfx1250) — matrix operand order repaired and checked by reading, not compiled, not run on hardware
//
#include <hip/hip_runtime.h>
#include <hip/hip_bf16.h>

// ---------------------------------------------------------------------------
// GCN-BERT classifier forward for MI455X (gfx1250, wave32, WMMA).
// B=64, S=512, D=1024, M=512, P=3.
// GEMMs on v_wmma_f32_16x16x32_f16, 128x128 block tile, double-buffered LDS.
// ---------------------------------------------------------------------------

#define B_   64
#define S_   512
#define D_   1024
#define M_   512
#define P_   3
#define LN_EPS 1e-6f

typedef __attribute__((ext_vector_type(16))) _Float16 v16h;
typedef __attribute__((ext_vector_type(8)))  float    v8f;

// ---------------- block reduction helper (wave32) ----------------
__device__ inline float block_reduce_sum(float x, float* smem /* >=8 floats */) {
    #pragma unroll
    for (int off = 16; off > 0; off >>= 1) x += __shfl_down(x, off, 32);
    const int lane = threadIdx.x & 31;
    const int w    = threadIdx.x >> 5;
    if (lane == 0) smem[w] = x;
    __syncthreads();
    if (w == 0) {
        x = (lane < 8) ? smem[lane] : 0.0f;
        #pragma unroll
        for (int off = 4; off > 0; off >>= 1) x += __shfl_down(x, off, 32);
        if (lane == 0) smem[0] = x;
    }
    __syncthreads();
    x = smem[0];
    __syncthreads();
    return x;
}

// ---------------- init: zero the loss accumulator ----------------
__global__ void init_kernel(float* loss_acc) { *loss_acc = 0.0f; }

// ---------------- LayerNorm + aspect head ----------------
// One block per (b,s) row of D=1024; each thread owns 4 contiguous floats.
__global__ __launch_bounds__(256)
void ln_aspect_kernel(const float* __restrict__ seq,
                      const int*   __restrict__ amask,
                      const float* __restrict__ ln_a,
                      const float* __restrict__ ln_b,
                      const float* __restrict__ asp_w,
                      const float* __restrict__ asp_b,
                      _Float16*    __restrict__ xh,
                      float*       __restrict__ aspect_pred,
                      float*       __restrict__ loss_acc) {
    __shared__ float sred[8];
    const int s   = blockIdx.x;
    const int b   = blockIdx.y;
    const long long row = (long long)b * S_ + s;
    const int tid = threadIdx.x;
    const int j0  = tid * 4;

    const float4 xv = *(const float4*)(seq + row * D_ + j0);
    const float v[4] = {xv.x, xv.y, xv.z, xv.w};

    float lsum = v[0] + v[1] + v[2] + v[3];
    const float mean = block_reduce_sum(lsum, sred) * (1.0f / D_);

    float lss = 0.0f;
    #pragma unroll
    for (int i = 0; i < 4; ++i) { const float c = v[i] - mean; lss += c * c; }
    const float ssq  = block_reduce_sum(lss, sred);
    const float stdv = sqrtf(ssq * (1.0f / (D_ - 1)));       // unbiased (Bessel)
    const float inv  = 1.0f / (stdv + LN_EPS);

    const float4 av = *(const float4*)(ln_a + j0);
    const float4 bv = *(const float4*)(ln_b + j0);
    const float4 wv = *(const float4*)(asp_w + j0);
    const float aa[4] = {av.x, av.y, av.z, av.w};
    const float bb[4] = {bv.x, bv.y, bv.z, bv.w};
    const float ww[4] = {wv.x, wv.y, wv.z, wv.w};

    union { _Float16 h[4]; uint2 u; } pk;
    float zpart = 0.0f;
    #pragma unroll
    for (int i = 0; i < 4; ++i) {
        const float xn = aa[i] * ((v[i] - mean) * inv) + bb[i];
        pk.h[i] = (_Float16)xn;
        zpart += xn * ww[i];
    }
    *(uint2*)(xh + row * D_ + j0) = pk.u;
    const float z = block_reduce_sum(zpart, sred) + asp_b[0];

    if (tid == 0) {
        const float tgt  = (float)amask[row];
        aspect_pred[row] = 1.0f / (1.0f + expf(-z));
        const float sp   = fmaxf(z, 0.0f) + log1pf(expf(-fabsf(z)));  // softplus
        atomicAdd(loss_acc, tgt * (sp - z * tgt));
    }
}

// ---------------- adjacency prep: f16 convert + reciprocal degree ------------
__global__ __launch_bounds__(256)
void adj_prep_kernel(const float* __restrict__ adj,
                     _Float16*    __restrict__ adj_h,
                     float*       __restrict__ rdenom) {
    __shared__ float sred[8];
    const int s = blockIdx.x;
    const int b = blockIdx.y;
    const long long base = ((long long)b * S_ + s) * S_;
    const int tid = threadIdx.x;
    const float2 a = *(const float2*)(adj + base + tid * 2);
    union { _Float16 h[2]; unsigned int u; } pk;
    pk.h[0] = (_Float16)a.x;
    pk.h[1] = (_Float16)a.y;
    *(unsigned int*)(adj_h + base + tid * 2) = pk.u;
    float sum = block_reduce_sum(a.x + a.y, sred);
    if (tid == 0) rdenom[(long long)b * S_ + s] = 1.0f / (sum + 1.0f);
}

// ---------------- fp32 -> f16 transposed convert (weights) ----------------
// dst[n*K + k] = (f16) src[k*N + n]
__global__ void f2h_t_kernel(const float* __restrict__ src, _Float16* __restrict__ dst,
                             int K, int N) {
    const int i = blockIdx.x * blockDim.x + threadIdx.x;
    if (i < K * N) {
        const int k = i / N, n = i - k * N;
        dst[n * K + k] = (_Float16)src[i];
    }
}

// ---------------- WMMA fragment loaders (LDS pitch = 40 halves) -------------
#define LDSP 40

// 16-bit A 16x32 layout (ISA 7.12.2): lanes 0-15 hold K 0-7 & 16-23,
// lanes 16-31 hold K 8-15 & 24-31, packed as pairs -> two b128 loads.
__device__ inline v16h frag_a(const _Float16* rowp, int kh) {
    v16h f;
    ((uint4*)&f)[0] = *(const uint4*)(rowp + 8 * kh);
    ((uint4*)&f)[1] = *(const uint4*)(rowp + 8 * kh + 16);
    return f;
}
// B 32x16: lanes 0-15 hold K 0-7 & 16-23, lanes 16-31 hold K 8-15 & 24-31,
// read from the n-major LDS tile (row = n, 32 k's).
__device__ inline v16h frag_b(const _Float16* rowp, int khi) {
    v16h f;
    ((uint4*)&f)[0] = *(const uint4*)(rowp + 8 * khi);
    ((uint4*)&f)[1] = *(const uint4*)(rowp + 8 * khi + 16);
    return f;
}

// ---------------- batched WMMA GEMM: C[b] = A[b] (512 x K) @ B[b] (K x N) ----
// Block tile 128x128, 8 waves as 4(M) x 2(N), wave tile 32x64 (2x4 WMMA).
// Double-buffered LDS, one barrier per K-step.
// BT==1: B is pre-transposed in memory ([N][K], n-major) -> straight copies.
// BT==0: B is row-major ([K][N]) -> transpose while staging (packed b32).
// EPI==0: C=f16(acc).  EPI==1: C=f16(relu((acc+bias[n])*rdenom[b,m])).
template <int EPI, int BT>
__global__ __launch_bounds__(256)
void wmma_gemm_kernel(const _Float16* __restrict__ A,  long long strideA,
                      const _Float16* __restrict__ Bm, long long strideB,
                      _Float16*       __restrict__ C,  long long strideC,
                      int N, int K,
                      const float* __restrict__ bias,
                      const float* __restrict__ rdenom) {
    __shared__ _Float16 sA[2][128 * LDSP];
    __shared__ _Float16 sB[2][128 * LDSP];   // row = n (0..127), 32 k's

    const int b        = blockIdx.z;
    const _Float16* gA = A  + (long long)b * strideA;
    const _Float16* gB = Bm + (long long)b * strideB;
    _Float16*       gC = C  + (long long)b * strideC;
    const int m_base   = blockIdx.y * 128;
    const int n_base   = blockIdx.x * 128;
    const int tid      = threadIdx.x;
    const int lane     = tid & 31;
    const int wave     = tid >> 5;
    const int wm0      = (wave & 3) * 32;
    const int wn0      = (wave >> 2) * 64;

    v8f acc[2][4] = {};

    // A staging: thread -> (row 0..127, 16-half chunk 0/16)
    const int ar = tid >> 1;
    const int ac = (tid & 1) * 16;
    // B staging, BT==0: thread -> (k-pair 0..15, n-chunk of 8)
    const int kp = tid >> 4;
    const int nc = (tid & 15) * 8;

    uint4 ra0, ra1, rb0, rb1;

    // ---- prefetch tile 0 into LDS buffer 0 ----
    {
        const _Float16* pA = gA + (long long)(m_base + ar) * K + ac;
        ra0 = ((const uint4*)pA)[0];
        ra1 = ((const uint4*)pA)[1];
        if (BT) {
            const _Float16* pB = gB + (long long)(n_base + ar) * K + ac;
            rb0 = ((const uint4*)pB)[0];
            rb1 = ((const uint4*)pB)[1];
        } else {
            const _Float16* pB = gB + (long long)(2 * kp) * N + n_base + nc;
            rb0 = *(const uint4*)pB;
            rb1 = *(const uint4*)(pB + N);
        }
        *(uint4*)&sA[0][ar * LDSP + ac]     = ra0;
        *(uint4*)&sA[0][ar * LDSP + ac + 8] = ra1;
        if (BT) {
            *(uint4*)&sB[0][ar * LDSP + ac]     = rb0;
            *(uint4*)&sB[0][ar * LDSP + ac + 8] = rb1;
        } else {
            const unsigned short* h0 = (const unsigned short*)&rb0;
            const unsigned short* h1 = (const unsigned short*)&rb1;
            #pragma unroll
            for (int i = 0; i < 8; ++i)
                *(unsigned int*)&sB[0][(nc + i) * LDSP + 2 * kp] =
                    (unsigned int)h0[i] | ((unsigned int)h1[i] << 16);
        }
    }
    __syncthreads();

    int cur = 0;
    const int am = lane & 15;
    const int kh = lane >> 4;

    for (int kb = 0; kb < K; kb += 32) {
        const bool has_next = (kb + 32) < K;
        // ---- issue next tile's global loads (wait lands after the WMMAs) ----
        if (has_next) {
            const int kn = kb + 32;
            const _Float16* pA = gA + (long long)(m_base + ar) * K + kn + ac;
            ra0 = ((const uint4*)pA)[0];
            ra1 = ((const uint4*)pA)[1];
            if (BT) {
                const _Float16* pB = gB + (long long)(n_base + ar) * K + kn + ac;
                rb0 = ((const uint4*)pB)[0];
                rb1 = ((const uint4*)pB)[1];
            } else {
                const _Float16* pB = gB + (long long)(kn + 2 * kp) * N + n_base + nc;
                rb0 = *(const uint4*)pB;
                rb1 = *(const uint4*)(pB + N);
            }
        }
        // ---- compute on current LDS buffer ----
        #pragma unroll
        for (int mt = 0; mt < 2; ++mt) {
            const v16h af = frag_a(&sA[cur][(wm0 + mt * 16 + am) * LDSP], kh);
            #pragma unroll
            for (int nt = 0; nt < 4; ++nt) {
                const v16h bf = frag_b(&sB[cur][(wn0 + nt * 16 + am) * LDSP], kh);
                acc[mt][nt] = __builtin_amdgcn_wmma_f32_16x16x32_f16(
                    false, af, false, bf, (short)0, acc[mt][nt], false, false);
            }
        }
        // ---- stage next tile into the other buffer ----
        if (has_next) {
            const int nxt = cur ^ 1;
            *(uint4*)&sA[nxt][ar * LDSP + ac]     = ra0;
            *(uint4*)&sA[nxt][ar * LDSP + ac + 8] = ra1;
            if (BT) {
                *(uint4*)&sB[nxt][ar * LDSP + ac]     = rb0;
                *(uint4*)&sB[nxt][ar * LDSP + ac + 8] = rb1;
            } else {
                const unsigned short* h0 = (const unsigned short*)&rb0;
                const unsigned short* h1 = (const unsigned short*)&rb1;
                #pragma unroll
                for (int i = 0; i < 8; ++i)
                    *(unsigned int*)&sB[nxt][(nc + i) * LDSP + 2 * kp] =
                        (unsigned int)h0[i] | ((unsigned int)h1[i] << 16);
            }
            __syncthreads();
            cur = nxt;
        }
    }

    // epilogue: C/D layout — element i: M = i + 8*(lane>>4), N = lane&15
    const int cn = lane & 15;
    const int mo = (lane >> 4) * 8;
    #pragma unroll
    for (int mt = 0; mt < 2; ++mt)
        #pragma unroll
        for (int nt = 0; nt < 4; ++nt)
            #pragma unroll
            for (int i = 0; i < 8; ++i) {
                const int row = m_base + wm0 + mt * 16 + mo + i;
                const int col = n_base + wn0 + nt * 16 + cn;
                float v = acc[mt][nt][i];
                if (EPI == 1) {
                    v = (v + bias[col]) * rdenom[(long long)b * S_ + row];
                    v = v > 0.0f ? v : 0.0f;
                }
                gC[(long long)row * N + col] = (_Float16)v;
            }
}

// ---------------- pooling: aspect-weighted mean + global mean ----------------
__global__ __launch_bounds__(256)
void pool_kernel(const _Float16* __restrict__ out,
                 const int*      __restrict__ amask,
                 float*          __restrict__ pooled) {
    const int b = blockIdx.y;
    const int m = blockIdx.x * 256 + threadIdx.x;
    const _Float16* p  = out + (long long)b * S_ * M_ + m;
    const int*      am = amask + (long long)b * S_;
    float dep = 0.0f, glob = 0.0f, wn = 0.0f;
    for (int s = 0; s < S_; ++s) {
        const float v = (float)p[(long long)s * M_];
        const float w = (float)am[s];
        dep  += w * v;
        glob += v;
        wn   += w;
    }
    pooled[(long long)b * 2 * M_ + m]      = dep / wn;
    pooled[(long long)b * 2 * M_ + M_ + m] = glob * (1.0f / S_);
}

// ---------------- classifier + softmax + loss finalize ----------------
__global__ __launch_bounds__(256)
void cls_kernel(const float* __restrict__ pooled,
                const float* __restrict__ cw,
                const float* __restrict__ cb,
                float* __restrict__ feat_out,
                float* __restrict__ logits_out,
                const float* __restrict__ loss_acc,
                float* __restrict__ loss_out) {
    __shared__ float sred[8];
    const int b = blockIdx.x;
    const float* pr = pooled + (long long)b * 2 * M_;
    float a0 = 0.0f, a1 = 0.0f, a2 = 0.0f;
    for (int j = threadIdx.x; j < 2 * M_; j += 256) {
        const float v = pr[j];
        a0 += v * cw[j * P_ + 0];
        a1 += v * cw[j * P_ + 1];
        a2 += v * cw[j * P_ + 2];
    }
    a0 = block_reduce_sum(a0, sred);
    a1 = block_reduce_sum(a1, sred);
    a2 = block_reduce_sum(a2, sred);
    if (threadIdx.x == 0) {
        const float f0 = a0 + cb[0], f1 = a1 + cb[1], f2 = a2 + cb[2];
        feat_out[b * P_ + 0] = f0;
        feat_out[b * P_ + 1] = f1;
        feat_out[b * P_ + 2] = f2;
        const float mx = fmaxf(f0, fmaxf(f1, f2));
        const float e0 = expf(f0 - mx), e1 = expf(f1 - mx), e2 = expf(f2 - mx);
        const float rs = 1.0f / (e0 + e1 + e2);
        logits_out[b * P_ + 0] = e0 * rs;
        logits_out[b * P_ + 1] = e1 * rs;
        logits_out[b * P_ + 2] = e2 * rs;
        if (b == 0) *loss_out = (*loss_acc) * (1.0f / ((float)B_ * (float)S_));
    }
}

// ---------------------------------------------------------------------------
extern "C" void kernel_launch(void* const* d_in, const int* in_sizes, int n_in,
                              void* d_out, int out_size, void* d_ws, size_t ws_size,
                              hipStream_t stream) {
    const float* seq    = (const float*)d_in[0];
    const float* adj    = (const float*)d_in[1];
    const int*   amask  = (const int*)  d_in[2];
    const float* ln_a   = (const float*)d_in[3];
    const float* ln_b   = (const float*)d_in[4];
    const float* asp_w  = (const float*)d_in[5];
    const float* asp_b  = (const float*)d_in[6];
    const float* W0     = (const float*)d_in[7];
    const float* b0     = (const float*)d_in[8];
    const float* W1     = (const float*)d_in[9];
    const float* b1     = (const float*)d_in[10];
    const float* W2     = (const float*)d_in[11];
    const float* b2     = (const float*)d_in[12];
    const float* cls_w  = (const float*)d_in[13];
    const float* cls_b  = (const float*)d_in[14];

    float* out        = (float*)d_out;
    float* feat_out   = out;                       // [64,3]
    float* logits_out = out + B_ * P_;             // [64,3]
    float* asp_out    = out + 2 * B_ * P_;         // [64,512]
    float* loss_out   = out + 2 * B_ * P_ + B_ * S_;

    // ---- workspace layout (bytes, all 256-aligned) ----
    char* ws = (char*)d_ws;
    constexpr size_t SZ_ADJ  = (size_t)B_ * S_ * S_ * 2;
    constexpr size_t SZ_ACTA = (size_t)B_ * S_ * D_ * 2;
    constexpr size_t SZ_AX   = (size_t)B_ * S_ * D_ * 2;
    constexpr size_t SZ_ACTB = (size_t)B_ * S_ * M_ * 2;
    constexpr size_t SZ_RD   = (size_t)B_ * S_ * 4;
    constexpr size_t SZ_W0   = (size_t)D_ * M_ * 2;
    constexpr size_t SZ_W1   = (size_t)M_ * M_ * 2;
    constexpr size_t SZ_W2   = (size_t)M_ * M_ * 2;

    size_t off = 0;
    _Float16* adj_h = (_Float16*)(ws + off); off += SZ_ADJ;
    _Float16* actA  = (_Float16*)(ws + off); off += SZ_ACTA;   // x, then layer-1 out
    _Float16* axbuf = (_Float16*)(ws + off); off += SZ_AX;
    _Float16* actB  = (_Float16*)(ws + off); off += SZ_ACTB;   // layer-0/2 out
    float*    rden  = (float*)   (ws + off); off += SZ_RD;
    _Float16* W0t   = (_Float16*)(ws + off); off += SZ_W0;     // [M][D] n-major
    _Float16* W1t   = (_Float16*)(ws + off); off += SZ_W1;     // [M][M]
    _Float16* W2t   = (_Float16*)(ws + off); off += SZ_W2;     // [M][M]
    float*    pooled = (float*)  (ws + off); off += (size_t)B_ * 2 * M_ * 4;
    float*    loss_acc = (float*)(ws + off); off += 256;

    // ---- pipeline ----
    init_kernel<<<1, 1, 0, stream>>>(loss_acc);

    ln_aspect_kernel<<<dim3(S_, B_), 256, 0, stream>>>(
        seq, amask, ln_a, ln_b, asp_w, asp_b, actA, asp_out, loss_acc);

    adj_prep_kernel<<<dim3(S_, B_), 256, 0, stream>>>(adj, adj_h, rden);

    f2h_t_kernel<<<(D_ * M_ + 255) / 256, 256, 0, stream>>>(W0, W0t, D_, M_);
    f2h_t_kernel<<<(M_ * M_ + 255) / 256, 256, 0, stream>>>(W1, W1t, M_, M_);
    f2h_t_kernel<<<(M_ * M_ + 255) / 256, 256, 0, stream>>>(W2, W2t, M_, M_);

    const long long sAdj = (long long)S_ * S_;
    const long long sXD  = (long long)S_ * D_;
    const long long sXM  = (long long)S_ * M_;

    // ---- layer 0: agg (512 x 1024, K=512) then proj (512 x 512, K=1024) ----
    wmma_gemm_kernel<0, 0><<<dim3(D_ / 128, S_ / 128, B_), 256, 0, stream>>>(
        adj_h, sAdj, actA, sXD, axbuf, sXD, D_, S_, nullptr, nullptr);
    wmma_gemm_kernel<1, 1><<<dim3(M_ / 128, S_ / 128, B_), 256, 0, stream>>>(
        axbuf, sXD, W0t, 0, actB, sXM, M_, D_, b0, rden);

    // ---- layer 1 ----
    wmma_gemm_kernel<0, 0><<<dim3(M_ / 128, S_ / 128, B_), 256, 0, stream>>>(
        adj_h, sAdj, actB, sXM, axbuf, sXM, M_, S_, nullptr, nullptr);
    wmma_gemm_kernel<1, 1><<<dim3(M_ / 128, S_ / 128, B_), 256, 0, stream>>>(
        axbuf, sXM, W1t, 0, actA, sXM, M_, M_, b1, rden);

    // ---- layer 2 ----
    wmma_gemm_kernel<0, 0><<<dim3(M_ / 128, S_ / 128, B_), 256, 0, stream>>>(
        adj_h, sAdj, actA, sXM, axbuf, sXM, M_, S_, nullptr, nullptr);
    wmma_gemm_kernel<1, 1><<<dim3(M_ / 128, S_ / 128, B_), 256, 0, stream>>>(
        axbuf, sXM, W2t, 0, actB, sXM, M_, M_, b2, rden);

    // ---- pooling + classifier ----
    pool_kernel<<<dim3(M_ / 256, B_), 256, 0, stream>>>(actB, amask, pooled);
    cls_kernel<<<B_, 256, 0, stream>>>(pooled, cls_w, cls_b,
                                       feat_out, logits_out, loss_acc, loss_out);

    (void)in_sizes; (void)n_in; (void)out_size; (void)ws_size;
}
